// MPNNDecoderVAE_39779987095907
// MI455X (gfx1250) — compile-verified, matrix operand order checked by reading, not run on hardware
//
#include <hip/hip_runtime.h>

// ---------------------------------------------------------------------------
// Types for CDNA5 WMMA (wave32): v16bf A/B fragments, v8f accumulator.
// ---------------------------------------------------------------------------
typedef __attribute__((ext_vector_type(16))) __bf16 bf16x16;
typedef __attribute__((ext_vector_type(8)))  __bf16 bf16x8;
typedef __attribute__((ext_vector_type(8)))  float  floatx8;

#define LORA_SCAL 2.0f   // alpha/r = 16/8

__device__ __forceinline__ unsigned short f2bf(float f) {
  unsigned int u = __float_as_uint(f);
  u += 0x7FFFu + ((u >> 16) & 1u);        // round-to-nearest-even
  return (unsigned short)(u >> 16);
}
__device__ __forceinline__ float bf2f(unsigned short h) {
  return __uint_as_float(((unsigned int)h) << 16);
}

// Fragment load for 16-bit WMMA operands, row-major source.
// p = row_base + kbase + 8*hi (16B aligned). Elements 0..7 <- K+0..7,
// elements 8..15 <- K+16..23 (the +8*hi half-wave offset covers the rest).
__device__ __forceinline__ bf16x16 frag_ld(const unsigned short* p) {
  bf16x8 lo = *reinterpret_cast<const bf16x8*>(p);
  bf16x8 hh = *reinterpret_cast<const bf16x8*>(p + 16);
  bf16x16 r;
#pragma unroll
  for (int i = 0; i < 8; ++i) { r[i] = lo[i]; r[i + 8] = hh[i]; }
  return r;
}

#define WMMA_BF16(a, b, c) \
  __builtin_amdgcn_wmma_f32_16x16x32_bf16(false, (a), false, (b), (short)0, (c), false, false)

// ---------------------------------------------------------------------------
// Prep kernels: LoRA effective weights + fp32 -> bf16 conversion
// ---------------------------------------------------------------------------
__global__ void cvt_bf16_kernel(const float* __restrict__ in,
                                unsigned short* __restrict__ out, int n) {
  int idx = blockIdx.x * blockDim.x + threadIdx.x;
  if (idx < n) out[idx] = f2bf(in[idx]);
}

// Weff = W + scal * B@A  for z_to_nodes: [16384,128], R=8
__global__ void prep_z2n_kernel(const float* __restrict__ w,
                                const float* __restrict__ A,
                                const float* __restrict__ Bm,
                                unsigned short* __restrict__ out) {
  int idx = blockIdx.x * blockDim.x + threadIdx.x;   // 2,097,152
  int o = idx >> 7, k = idx & 127;
  float acc = w[idx];
#pragma unroll
  for (int r = 0; r < 8; ++r) acc += LORA_SCAL * Bm[o * 8 + r] * A[r * 128 + k];
  out[idx] = f2bf(acc);
}

// node predictor Weff: [64,256]
__global__ void prep_node_kernel(const float* __restrict__ w,
                                 const float* __restrict__ A,
                                 const float* __restrict__ Bm,
                                 unsigned short* __restrict__ out) {
  int idx = blockIdx.x * blockDim.x + threadIdx.x;   // 16384
  int o = idx >> 8, k = idx & 255;
  float acc = w[idx];
#pragma unroll
  for (int r = 0; r < 8; ++r) acc += LORA_SCAL * Bm[o * 8 + r] * A[r * 256 + k];
  out[idx] = f2bf(acc);
}

// adj Weff [512] followed by bond Weff [4*512], fp32 (tiny, reused a lot)
__global__ void prep_ab_kernel(const float* __restrict__ aw, const float* __restrict__ aA,
                               const float* __restrict__ aB, const float* __restrict__ bw,
                               const float* __restrict__ bA, const float* __restrict__ bB,
                               float* __restrict__ wab) {
  int idx = blockIdx.x * blockDim.x + threadIdx.x;   // 2560
  if (idx < 512) {
    float acc = aw[idx];
#pragma unroll
    for (int r = 0; r < 8; ++r) acc += LORA_SCAL * aB[r] * aA[r * 512 + idx];
    wab[idx] = acc;
  } else if (idx < 2560) {
    int t = idx - 512; int o = t >> 9, k = t & 511;
    float acc = bw[t];
#pragma unroll
    for (int r = 0; r < 8; ++r) acc += LORA_SCAL * bB[o * 8 + r] * bA[r * 512 + k];
    wab[idx] = acc;
  }
}

// ---------------------------------------------------------------------------
// GEMM 1: x = relu(z @ Weff^T + b)   M=512, N=16384, K=128 -> xbuf bf16
// One wave per 16x64 strip (A fragment reused across 4 N-tiles).
// Strips: 32 (M) x 256 (N) = 8192 waves.
// ---------------------------------------------------------------------------
__global__ __launch_bounds__(256)
void gemm_z2n_kernel(const unsigned short* __restrict__ zb,
                     const unsigned short* __restrict__ w,
                     const float* __restrict__ bias,
                     unsigned short* __restrict__ x) {
  int wid = blockIdx.x * 8 + (threadIdx.x >> 5);
  int tM = wid >> 8, tN = wid & 255;
  int lane = threadIdx.x & 31, lr = lane & 15, hi = lane >> 4;
  const unsigned short* ar = zb + (tM * 16 + lr) * 128 + 8 * hi;
  const unsigned short* br = w  + (tN * 64 + lr) * 128 + 8 * hi;
  floatx8 acc[4] = {};
#pragma unroll
  for (int kb = 0; kb < 128; kb += 32) {
    bf16x16 a = frag_ld(ar + kb);
#pragma unroll
    for (int t = 0; t < 4; ++t)
      acc[t] = WMMA_BF16(a, frag_ld(br + t * (16 * 128) + kb), acc[t]);
  }
#pragma unroll
  for (int t = 0; t < 4; ++t) {
    int col = tN * 64 + t * 16 + lr;
    float bv = bias[col];
#pragma unroll
    for (int r = 0; r < 8; ++r) {
      int m = tM * 16 + r + 8 * hi;
      float v = fmaxf(acc[t][r] + bv, 0.f);
      x[m * 16384 + col] = f2bf(v);   // [B, N*H] flat == [B*N, H] flat
    }
  }
}

// ---------------------------------------------------------------------------
// GEMM 2 (per layer): h = concat(x, z) @ gatW[l]^T
// M=32768, N=256, K=384.  Wave computes 16x64 strip; 64-col group == 1 head.
// Output stored transposed per (b,head):
//   hT[((b*4+hd)*64 + c)*64 + j]   (c = channel, j = source node)
// Strips: 2048 (M) x 4 (head) = 8192 waves.
// ---------------------------------------------------------------------------
__global__ __launch_bounds__(256)
void gemm_xc_kernel(const unsigned short* __restrict__ x,
                    const unsigned short* __restrict__ zb,
                    const unsigned short* __restrict__ gatw,
                    unsigned short* __restrict__ hT, int layer) {
  int wid = blockIdx.x * 8 + (threadIdx.x >> 5);
  int tM = wid >> 2, g = wid & 3;                    // g == head
  int lane = threadIdx.x & 31, lr = lane & 15, hi = lane >> 4;
  int mA = tM * 16 + lr;
  const unsigned short* xr = x  + mA * 256 + 8 * hi;
  const unsigned short* zr = zb + (mA >> 6) * 128 + 8 * hi;
  const unsigned short* wr = gatw + layer * (256 * 384) + (g * 64 + lr) * 384 + 8 * hi;
  floatx8 acc[4] = {};
#pragma unroll
  for (int kb = 0; kb < 384; kb += 32) {
    bf16x16 a = (kb < 256) ? frag_ld(xr + kb) : frag_ld(zr + (kb - 256));
#pragma unroll
    for (int t = 0; t < 4; ++t)
      acc[t] = WMMA_BF16(a, frag_ld(wr + t * (16 * 384) + kb), acc[t]);
  }
#pragma unroll
  for (int t = 0; t < 4; ++t) {
    int c = t * 16 + lr;
#pragma unroll
    for (int r = 0; r < 8; ++r) {
      int m = tM * 16 + r + 8 * hi;
      int b = m >> 6, j = m & 63;
      hT[(((b * 4 + g) * 64 + c) * 64) + j] = f2bf(acc[t][r]);
    }
  }
}

// e_src / e_dst: per (b,head,node) dot of h with attention vectors
__global__ void esed_kernel(const unsigned short* __restrict__ hT,
                            const float* __restrict__ asrc,
                            const float* __restrict__ adst,
                            float* __restrict__ es, float* __restrict__ ed, int layer) {
  int idx = blockIdx.x * blockDim.x + threadIdx.x;   // 131072 = B*4*64
  int j = idx & 63, bh = idx >> 6, hd = bh & 3;
  const unsigned short* base = hT + bh * 4096 + j;   // stride 64 over channels
  const float* as = asrc + layer * 256 + hd * 64;
  const float* ad = adst + layer * 256 + hd * 64;
  float s = 0.f, d = 0.f;
  for (int c = 0; c < 64; ++c) {
    float v = bf2f(base[c * 64]);
    s += v * as[c];
    d += v * ad[c];
  }
  es[idx] = s; ed[idx] = d;
}

// alpha[i,j] = softmax_j( leaky_relu(ed[i] + es[j]) ), written bf16
__global__ void softmax_kernel(const float* __restrict__ es,
                               const float* __restrict__ ed,
                               unsigned short* __restrict__ alpha) {
  int idx = blockIdx.x * blockDim.x + threadIdx.x;   // 131072 = B*4*64 rows
  int i = idx & 63, bh = idx >> 6;
  const float* esr = es + bh * 64;
  float di = ed[bh * 64 + i];
  float mx = -1e30f;
  for (int j = 0; j < 64; ++j) {
    float e = di + esr[j]; e = e > 0.f ? e : 0.2f * e;
    mx = fmaxf(mx, e);
  }
  float sum = 0.f;
  for (int j = 0; j < 64; ++j) {
    float e = di + esr[j]; e = e > 0.f ? e : 0.2f * e;
    sum += __expf(e - mx);
  }
  float rs = 1.f / sum;
  unsigned short* arow = alpha + (bh * 64 + i) * 64;
  for (int j = 0; j < 64; ++j) {
    float e = di + esr[j]; e = e > 0.f ? e : 0.2f * e;
    arow[j] = f2bf(__expf(e - mx) * rs);
  }
}

// ---------------------------------------------------------------------------
// GEMM 3 (per layer): x_new = relu(alpha @ h + bias). Per (b,head): 64x64x64.
// Wave computes 16(i) x 64(c); A = alpha row block, B = hT (contig along j).
// Waves: 2048 (b,head) x 4 (i-tiles) = 8192.
// ---------------------------------------------------------------------------
__global__ __launch_bounds__(256)
void gemm_attn_kernel(const unsigned short* __restrict__ alpha,
                      const unsigned short* __restrict__ hT,
                      const float* __restrict__ gbias,
                      unsigned short* __restrict__ x, int layer) {
  int wid = blockIdx.x * 8 + (threadIdx.x >> 5);
  int bh = wid >> 2, tI = wid & 3;
  int lane = threadIdx.x & 31, lr = lane & 15, hi = lane >> 4;
  const unsigned short* ar = alpha + bh * 4096 + (tI * 16 + lr) * 64 + 8 * hi;
  const unsigned short* br = hT    + bh * 4096 + lr * 64 + 8 * hi;
  floatx8 acc[4] = {};
#pragma unroll
  for (int kb = 0; kb < 64; kb += 32) {
    bf16x16 a = frag_ld(ar + kb);
#pragma unroll
    for (int t = 0; t < 4; ++t)
      acc[t] = WMMA_BF16(a, frag_ld(br + t * (16 * 64) + kb), acc[t]);
  }
  int b = bh >> 2, hd = bh & 3;
#pragma unroll
  for (int t = 0; t < 4; ++t) {
    int c = t * 16 + lr;
    float bv = gbias[layer * 256 + hd * 64 + c];
#pragma unroll
    for (int r = 0; r < 8; ++r) {
      int i = tI * 16 + r + 8 * hi;
      float v = fmaxf(acc[t][r] + bv, 0.f);
      x[(b * 64 + i) * 256 + hd * 64 + c] = f2bf(v);
    }
  }
}

// ---------------------------------------------------------------------------
// GEMM 4: recon = x @ nodeWeff^T + node_b.  M=32768, N=64, K=256, fp32 out.
// Wave computes full 16x64 strip; 2048 waves.
// ---------------------------------------------------------------------------
__global__ __launch_bounds__(256)
void gemm_node_kernel(const unsigned short* __restrict__ x,
                      const unsigned short* __restrict__ nodew,
                      const float* __restrict__ nodeb,
                      float* __restrict__ out) {
  int wid = blockIdx.x * 8 + (threadIdx.x >> 5);   // 2048 waves
  int tM = wid;
  int lane = threadIdx.x & 31, lr = lane & 15, hi = lane >> 4;
  const unsigned short* ar = x     + (tM * 16 + lr) * 256 + 8 * hi;
  const unsigned short* br = nodew + lr * 256 + 8 * hi;
  floatx8 acc[4] = {};
#pragma unroll
  for (int kb = 0; kb < 256; kb += 32) {
    bf16x16 a = frag_ld(ar + kb);
#pragma unroll
    for (int t = 0; t < 4; ++t)
      acc[t] = WMMA_BF16(a, frag_ld(br + t * (16 * 256) + kb), acc[t]);
  }
#pragma unroll
  for (int t = 0; t < 4; ++t) {
    int col = t * 16 + lr;
    float bv = nodeb[col];
#pragma unroll
    for (int r = 0; r < 8; ++r) {
      int m = tM * 16 + r + 8 * hi;
      out[m * 64 + col] = acc[t][r] + bv;
    }
  }
}

// Per-node dots for pairwise heads: s1,s2 (adj) and t1[4],t2[4] (bond)
__global__ void st_kernel(const unsigned short* __restrict__ x,
                          const float* __restrict__ wab,
                          float* __restrict__ st) {
  int m = blockIdx.x * blockDim.x + threadIdx.x;   // 32768
  const float* wa = wab;         // 512
  const float* wb = wab + 512;   // 4 x 512
  const unsigned short* xr = x + m * 256;
  float s1 = 0.f, s2 = 0.f, t1[4] = {0,0,0,0}, t2[4] = {0,0,0,0};
  for (int k = 0; k < 256; ++k) {
    float xv = bf2f(xr[k]);
    s1 += xv * wa[k];
    s2 += xv * wa[256 + k];
#pragma unroll
    for (int c = 0; c < 4; ++c) {
      t1[c] += xv * wb[c * 512 + k];
      t2[c] += xv * wb[c * 512 + 256 + k];
    }
  }
  float* o = st + m * 10;
  o[0] = s1; o[1] = s2;
#pragma unroll
  for (int c = 0; c < 4; ++c) { o[2 + c] = t1[c]; o[6 + c] = t2[c]; }
}

// Assemble triu/tril adj + bond logits
__global__ void assemble_kernel(const float* __restrict__ st,
                                const float* __restrict__ adjb,
                                const float* __restrict__ bondb,
                                float* __restrict__ adj_out,
                                float* __restrict__ bond_out) {
  int idx = blockIdx.x * blockDim.x + threadIdx.x;   // 2,097,152 = B*64*64
  int j = idx & 63, i = (idx >> 6) & 63, b = idx >> 12;
  int mi = b * 64 + i, mj = b * 64 + j;
  float a = 0.f, bo[4] = {0,0,0,0};
  if (j > i) {
    a = st[mi * 10] + st[mj * 10 + 1] + adjb[0];
#pragma unroll
    for (int c = 0; c < 4; ++c) bo[c] = st[mi * 10 + 2 + c] + st[mj * 10 + 6 + c] + bondb[c];
  } else if (i > j) {
    a = st[mj * 10] + st[mi * 10 + 1] + adjb[0];
#pragma unroll
    for (int c = 0; c < 4; ++c) bo[c] = st[mj * 10 + 2 + c] + st[mi * 10 + 6 + c] + bondb[c];
  }
  adj_out[idx] = a;
#pragma unroll
  for (int c = 0; c < 4; ++c) bond_out[idx * 4 + c] = bo[c];
}

// ---------------------------------------------------------------------------
// Launch
// ---------------------------------------------------------------------------
extern "C" void kernel_launch(void* const* d_in, const int* in_sizes, int n_in,
                              void* d_out, int out_size, void* d_ws, size_t ws_size,
                              hipStream_t stream) {
  (void)in_sizes; (void)n_in; (void)out_size; (void)ws_size;
  const float* z       = (const float*)d_in[0];
  const float* z2n_w   = (const float*)d_in[1];
  const float* z2n_b   = (const float*)d_in[2];
  const float* z2n_A   = (const float*)d_in[3];
  const float* z2n_B   = (const float*)d_in[4];
  const float* gat_W   = (const float*)d_in[5];
  const float* gat_as  = (const float*)d_in[6];
  const float* gat_ad  = (const float*)d_in[7];
  const float* gat_bias= (const float*)d_in[8];
  const float* node_w  = (const float*)d_in[9];
  const float* node_b  = (const float*)d_in[10];
  const float* node_A  = (const float*)d_in[11];
  const float* node_B  = (const float*)d_in[12];
  const float* adj_w   = (const float*)d_in[13];
  const float* adj_b   = (const float*)d_in[14];
  const float* adj_A   = (const float*)d_in[15];
  const float* adj_B   = (const float*)d_in[16];
  const float* bond_w  = (const float*)d_in[17];
  const float* bond_b  = (const float*)d_in[18];
  const float* bond_A  = (const float*)d_in[19];
  const float* bond_B  = (const float*)d_in[20];

  char* ws = (char*)d_ws;
  unsigned short* zb    = (unsigned short*)(ws);                       // 65,536
  unsigned short* wz2n  = zb + 65536;                                  // 2,097,152
  unsigned short* gatw  = wz2n + 2097152;                              // 294,912
  unsigned short* nodew = gatw + 294912;                               // 16,384
  float*          wab   = (float*)(ws + 4947968);                      // 2,560
  unsigned short* xbuf  = (unsigned short*)(ws + 4958208);             // 8,388,608
  unsigned short* hT    = (unsigned short*)(ws + 21735424);            // 8,388,608
  float*          es    = (float*)(ws + 38512640);                     // 131,072
  float*          ed    = (float*)(ws + 39036928);                     // 131,072
  unsigned short* alpha = (unsigned short*)(ws + 39561216);            // 8,388,608
  float*          st    = (float*)(ws + 56338432);                     // 327,680

  float* recon = (float*)d_out;
  float* adjo  = recon + 2097152;
  float* bondo = adjo + 2097152;

  // Prep: bf16 conversions + LoRA effective weights
  cvt_bf16_kernel<<<256, 256, 0, stream>>>(z, zb, 65536);
  prep_z2n_kernel<<<8192, 256, 0, stream>>>(z2n_w, z2n_A, z2n_B, wz2n);
  cvt_bf16_kernel<<<1152, 256, 0, stream>>>(gat_W, gatw, 294912);
  prep_node_kernel<<<64, 256, 0, stream>>>(node_w, node_A, node_B, nodew);
  prep_ab_kernel<<<10, 256, 0, stream>>>(adj_w, adj_A, adj_B, bond_w, bond_A, bond_B, wab);

  // z -> node features (WMMA GEMM, fused bias+ReLU), 8192 waves
  gemm_z2n_kernel<<<1024, 256, 0, stream>>>(zb, wz2n, z2n_b, xbuf);

  // 3 GAT layers
  for (int l = 0; l < 3; ++l) {
    gemm_xc_kernel<<<1024, 256, 0, stream>>>(xbuf, zb, gatw, hT, l);
    esed_kernel<<<512, 256, 0, stream>>>(hT, gat_as, gat_ad, es, ed, l);
    softmax_kernel<<<512, 256, 0, stream>>>(es, ed, alpha);
    gemm_attn_kernel<<<1024, 256, 0, stream>>>(alpha, hT, gat_bias, xbuf, l);
  }

  // Heads
  gemm_node_kernel<<<256, 256, 0, stream>>>(xbuf, nodew, node_b, recon);
  st_kernel<<<128, 256, 0, stream>>>(xbuf, wab, st);
  assemble_kernel<<<8192, 256, 0, stream>>>(st, adj_b, bond_b, adjo, bondo);
}
